// QKVAttention_687194768356
// MI455X (gfx1250) — matrix operand order repaired and checked by reading, not compiled, not run on hardware
//
#include <hip/hip_runtime.h>
#include <math.h>

// ---- CDNA5 WMMA vector types (probe-confirmed signatures) ----
typedef _Float16 h4   __attribute__((ext_vector_type(4)));
typedef _Float16 h8   __attribute__((ext_vector_type(8)));
typedef _Float16 v16h __attribute__((ext_vector_type(16)));
typedef float    f4   __attribute__((ext_vector_type(4)));
typedef float    v8f  __attribute__((ext_vector_type(8)));

static constexpr int BH       = 32;    // batch*heads
static constexpr int CH       = 64;    // channels per head
static constexpr int TLEN     = 2048;
static constexpr int SLEN     = 2048;
static constexpr int BM       = 64;    // T rows per workgroup
static constexpr int BN       = 64;    // S cols per iteration
static constexpr int NTHREADS = 128;   // 4 waves of 32
static constexpr int LDH      = 72;    // padded half-stride (144B, 16B-aligned rows)

// Fold both 1/sqrt(ch) factors AND log2(e) into the Q pre-scale: scores land in
// the exp2 domain, so softmax uses raw v_exp_f32 with no per-element multiply.
static constexpr float QSCALE = (1.0f / 64.0f) * 1.44269504088896340736f;

__device__ __forceinline__ v16h cat8(h8 lo, h8 hi) {
  return __builtin_shufflevector(lo, hi, 0,1,2,3,4,5,6,7,8,9,10,11,12,13,14,15);
}

// Raw max ops (scores are finite: skip llvm.maxnum's sNaN canonicalization).
// Non-volatile: pure register compute, scheduler may interleave across rows.
__device__ __forceinline__ float amax(float a, float b) {
  float d;
  asm("v_max_num_f32_e32 %0, %1, %2" : "=v"(d) : "v"(a), "v"(b));
  return d;
}
__device__ __forceinline__ float max4(float a, float b, float c, float d) {
  float x;
  asm("v_max3_num_f32 %0, %1, %2, %3" : "=v"(x) : "v"(a), "v"(b), "v"(c));
  return amax(x, d);
}

// Single-instruction 16-lane butterfly steps: VOP2 + DPP16 modifier.
// src0 reads through the DPP lane-crossbar, src1 is the plain register:
// x = op(dpp(x), x) in one VALU instruction, no LDS pipe traffic.
__device__ __forceinline__ float red_max16(float x) {
  asm("v_max_num_f32_dpp %0, %0, %0 quad_perm:[1,0,3,2] row_mask:0xf bank_mask:0xf bound_ctrl:1" : "+v"(x));
  asm("v_max_num_f32_dpp %0, %0, %0 quad_perm:[2,3,0,1] row_mask:0xf bank_mask:0xf bound_ctrl:1" : "+v"(x));
  asm("v_max_num_f32_dpp %0, %0, %0 row_half_mirror row_mask:0xf bank_mask:0xf bound_ctrl:1" : "+v"(x));
  asm("v_max_num_f32_dpp %0, %0, %0 row_mirror row_mask:0xf bank_mask:0xf bound_ctrl:1" : "+v"(x));
  return x;
}
__device__ __forceinline__ float red_sum16(float x) {
  asm("v_add_f32_dpp %0, %0, %0 quad_perm:[1,0,3,2] row_mask:0xf bank_mask:0xf bound_ctrl:1" : "+v"(x));
  asm("v_add_f32_dpp %0, %0, %0 quad_perm:[2,3,0,1] row_mask:0xf bank_mask:0xf bound_ctrl:1" : "+v"(x));
  asm("v_add_f32_dpp %0, %0, %0 row_half_mirror row_mask:0xf bank_mask:0xf bound_ctrl:1" : "+v"(x));
  asm("v_add_f32_dpp %0, %0, %0 row_mirror row_mask:0xf bank_mask:0xf bound_ctrl:1" : "+v"(x));
  return x;
}

__global__ __launch_bounds__(NTHREADS, 4)   // cap VGPRs <= 256: no MSB churn, 4 waves/SIMD
void qkv_attn_wmma_kernel(const float* __restrict__ q,
                          const float* __restrict__ k,
                          const float* __restrict__ v,
                          float* __restrict__ out) {
  // Manual LDS partition (halves): Qs | K0 | K1 | V0 | V1 | Ps   = 6 * 4608 halves = 55296 B
  // Os (f32, 64x65 = 16640 B) overlaps the K0/K1 region after the main loop.
  __shared__ __align__(16) unsigned char smem[6 * BM * LDH * 2];
  _Float16* const Qs  = (_Float16*)smem;                   // [t][c] * QSCALE
  _Float16* const Kb0 = Qs  + BM * LDH;                    // K^T tiles [s][c]
  _Float16* const Kb1 = Kb0 + BM * LDH;
  _Float16* const Vb0 = Kb1 + BM * LDH;                    // V tiles [c][s]
  _Float16* const Vb1 = Vb0 + BM * LDH;
  _Float16* const Ps  = Vb1 + BM * LDH;                    // per-wave P [t][s]
  float*    const Os  = (float*)(smem + 2 * BM * LDH);     // [t][c+pad] staging

  const int tid  = threadIdx.x;
  const int lane = tid & 31;
  const int wave = tid >> 5;
  const int g    = lane >> 4;   // ISA half-wave group for WMMA layouts
  const int ln   = lane & 15;

  const int bh = blockIdx.y;
  const int t0 = blockIdx.x * BM;

  const float* qb = q + (size_t)bh * CH * TLEN;
  const float* kb = k + (size_t)bh * CH * SLEN;
  const float* vb = v + (size_t)bh * CH * SLEN;
  float*       ob = out + (size_t)bh * CH * TLEN;

  // ---- Stage Q tile (vectorized loads, transposed b16 stores): Qs[t][c] = q[c][t0+t]*QSCALE
  #pragma unroll
  for (int j = 0; j < 8; ++j) {
    int ci = tid + j * NTHREADS;            // 1024 float4 chunks
    int c  = ci >> 4;
    int t4 = (ci & 15) * 4;
    f4 qv = *(const f4*)&qb[(size_t)c * TLEN + t0 + t4];
    #pragma unroll
    for (int e = 0; e < 4; ++e)
      Qs[(t4 + e) * LDH + c] = (_Float16)(qv[e] * QSCALE);
  }
  __syncthreads();

  // ---- Persistent A fragments of Q^T (16x32 f16: halves 0-7 -> K=g*8+j, 8-15 -> K=16+g*8+j)
  const int qrow = wave * 16 + ln;
  v16h aq0, aq1;
  {
    h8 lo0 = *(const h8*)&Qs[qrow * LDH + g * 8];
    h8 hi0 = *(const h8*)&Qs[qrow * LDH + 16 + g * 8];
    aq0 = cat8(lo0, hi0);
    h8 lo1 = *(const h8*)&Qs[qrow * LDH + 32 + g * 8];
    h8 hi1 = *(const h8*)&Qs[qrow * LDH + 48 + g * 8];
    aq1 = cat8(lo1, hi1);
  }

  v8f acc[4];
  #pragma unroll
  for (int n = 0; n < 4; ++n)
    acc[n] = (v8f){0.f,0.f,0.f,0.f,0.f,0.f,0.f,0.f};
  float mrow[8], lrow[8];
  #pragma unroll
  for (int r = 0; r < 8; ++r) { mrow[r] = -1e30f; lrow[r] = 0.0f; }

  // ---- Double-buffered pipeline: raw f32 tile in regs -> LDS(p) -> barrier -> next loads -> compute
  f4 kreg[8], vreg[8];
  #pragma unroll
  for (int j = 0; j < 8; ++j) {          // prologue: tile 0
    int ci = tid + j * NTHREADS;
    int c  = ci >> 4;
    int s4 = (ci & 15) * 4;
    kreg[j] = *(const f4*)&kb[(size_t)c * SLEN + s4];
    vreg[j] = *(const f4*)&vb[(size_t)c * SLEN + s4];
  }

  constexpr int NT = SLEN / BN;
  for (int it = 0; it < NT; ++it) {
    _Float16* const Kts = (it & 1) ? Kb1 : Kb0;
    _Float16* const Vs  = (it & 1) ? Vb1 : Vb0;

    // store staged tile (f32->f16 convert happens here, after a full tile of compute)
    #pragma unroll
    for (int j = 0; j < 8; ++j) {
      int ci = tid + j * NTHREADS;
      int c  = ci >> 4;
      int s4 = (ci & 15) * 4;
      h4 vh;
      #pragma unroll
      for (int e = 0; e < 4; ++e) {
        Kts[(s4 + e) * LDH + c] = (_Float16)kreg[j][e];   // transpose scatter
        vh[e] = (_Float16)vreg[j][e];
      }
      *(h4*)&Vs[c * LDH + s4] = vh;                        // contiguous b64
    }
    __syncthreads();

    // issue next tile's global loads; loadcnt wait lands in next iteration's store phase
    if (it + 1 < NT) {
      int s0n = (it + 1) * BN;
      #pragma unroll
      for (int j = 0; j < 8; ++j) {
        int ci = tid + j * NTHREADS;
        int c  = ci >> 4;
        int s4 = (ci & 15) * 4;
        kreg[j] = *(const f4*)&kb[(size_t)c * SLEN + s0n + s4];
        vreg[j] = *(const f4*)&vb[(size_t)c * SLEN + s0n + s4];
      }
    }

    // ---- Scores: sc[nt] = Q^T(16x64) x K(64 x 16-col tile), K-dim split 2x32
    v8f sc[4];
    #pragma unroll
    for (int nt = 0; nt < 4; ++nt) {
      int srow = nt * 16 + ln;  // B layout: N=ln, halves 0-7 -> K = g*8 + i, 8-15 -> K = 16 + g*8 + i
      h8 lo0 = *(const h8*)&Kts[srow * LDH + g * 8];
      h8 hi0 = *(const h8*)&Kts[srow * LDH + 16 + g * 8];
      v16h b0 = cat8(lo0, hi0);
      h8 lo1 = *(const h8*)&Kts[srow * LDH + 32 + g * 8];
      h8 hi1 = *(const h8*)&Kts[srow * LDH + 48 + g * 8];
      v16h b1 = cat8(lo1, hi1);
      v8f z = (v8f){0.f,0.f,0.f,0.f,0.f,0.f,0.f,0.f};
      z = __builtin_amdgcn_wmma_f32_16x16x32_f16(false, aq0, false, b0, (short)0, z, false, false);
      z = __builtin_amdgcn_wmma_f32_16x16x32_f16(false, aq1, false, b1, (short)0, z, false, false);
      sc[nt] = z;
    }

    // ---- Online softmax in exp2 domain (row M = r + 8g, col N = ln); DPP16 butterflies
    #pragma unroll
    for (int r = 0; r < 8; ++r) {
      float x = red_max16(max4(sc[0][r], sc[1][r], sc[2][r], sc[3][r]));
      float mnew  = amax(mrow[r], x);
      float alpha = __builtin_amdgcn_exp2f(mrow[r] - mnew);
      mrow[r] = mnew;
      lrow[r] *= alpha;
      #pragma unroll
      for (int n = 0; n < 4; ++n) acc[n][r] *= alpha;
    }

    // ---- P = exp2(sc - m); row sums; stash P (f16) for A-layout reload
    float rs[8];
    #pragma unroll
    for (int r = 0; r < 8; ++r) rs[r] = 0.0f;
    #pragma unroll
    for (int nt = 0; nt < 4; ++nt) {
      #pragma unroll
      for (int r = 0; r < 8; ++r) {
        float p = __builtin_amdgcn_exp2f(sc[nt][r] - mrow[r]);
        rs[r] += p;
        Ps[(wave * 16 + r + 8 * g) * LDH + nt * 16 + ln] = (_Float16)p;
      }
    }
    #pragma unroll
    for (int r = 0; r < 8; ++r) lrow[r] += red_sum16(rs[r]);

    // ---- A fragments of P (same-wave LDS ops are in-order)
    const int prow = (wave * 16 + ln) * LDH;
    h8 plo0 = *(const h8*)&Ps[prow + g * 8];
    h8 phi0 = *(const h8*)&Ps[prow + 16 + g * 8];
    v16h ap0 = cat8(plo0, phi0);
    h8 plo1 = *(const h8*)&Ps[prow + 32 + g * 8];
    h8 phi1 = *(const h8*)&Ps[prow + 48 + g * 8];
    v16h ap1 = cat8(plo1, phi1);

    // ---- acc += P(16x64) x V^T(64 x 16-col tile of c)
    #pragma unroll
    for (int n = 0; n < 4; ++n) {
      int c = n * 16 + ln;  // B layout: N=ln (channel), halves 0-7 -> K = s = g*8 + i, 8-15 -> K = 16 + g*8 + i
      h8 lo0 = *(const h8*)&Vs[c * LDH + g * 8];
      h8 hi0 = *(const h8*)&Vs[c * LDH + 16 + g * 8];
      v16h b0 = cat8(lo0, hi0);
      h8 lo1 = *(const h8*)&Vs[c * LDH + 32 + g * 8];
      h8 hi1 = *(const h8*)&Vs[c * LDH + 48 + g * 8];
      v16h b1 = cat8(lo1, hi1);
      acc[n] = __builtin_amdgcn_wmma_f32_16x16x32_f16(false, ap0, false, b0, (short)0, acc[n], false, false);
      acc[n] = __builtin_amdgcn_wmma_f32_16x16x32_f16(false, ap1, false, b1, (short)0, acc[n], false, false);
    }
  }

  // ---- Finalize: out = acc / l; stage via LDS (overlaps dead K buffers); coalesced b128 stores
  __syncthreads();   // all waves done reading K/V buffers before Os overwrites them
  float linv[8];
  #pragma unroll
  for (int r = 0; r < 8; ++r) linv[r] = 1.0f / lrow[r];
  #pragma unroll
  for (int n = 0; n < 4; ++n) {
    #pragma unroll
    for (int r = 0; r < 8; ++r)
      Os[(wave * 16 + r + 8 * g) * 65 + n * 16 + ln] = acc[n][r] * linv[r];
  }
  __syncthreads();
  #pragma unroll
  for (int j = 0; j < 8; ++j) {
    int ci = tid + j * NTHREADS;
    int c  = ci >> 4;
    int t4 = (ci & 15) * 4;
    f4 o;
    #pragma unroll
    for (int e = 0; e < 4; ++e) o[e] = Os[(t4 + e) * 65 + c];
    *(f4*)&ob[(size_t)c * TLEN + t0 + t4] = o;
  }
}

extern "C" void kernel_launch(void* const* d_in, const int* in_sizes, int n_in,
                              void* d_out, int out_size, void* d_ws, size_t ws_size,
                              hipStream_t stream) {
  (void)in_sizes; (void)n_in; (void)out_size; (void)d_ws; (void)ws_size;
  const float* q = (const float*)d_in[0];
  const float* k = (const float*)d_in[1];
  const float* v = (const float*)d_in[2];
  float* out = (float*)d_out;
  dim3 grid(TLEN / BM, BH);
  dim3 block(NTHREADS);
  qkv_attn_wmma_kernel<<<grid, block, 0, stream>>>(q, k, v, out);
}
